// MultiHeadLatentAttention_7662221656423
// MI455X (gfx1250) — matrix operand order repaired and checked by reading, not compiled, not run on hardware
//
#include <hip/hip_runtime.h>
#include <hip/hip_bf16.h>

// ---------------------------------------------------------------------------
// MLA forward for MI455X (gfx1250, wave32, WMMA).
// ~216 GFLOP of matmul vs ~0.3 GB HBM traffic -> matrix-core bound: every
// matmul runs on V_WMMA_F32_16X16X32_F16 (f32 accumulate).
//  * Weights are converted f32->f16 *transposed* (N-major) so every WMMA
//    B-fragment is a per-lane contiguous 128-bit load.
//  * GEMM tiles are staged with gfx1250 async direct-to-LDS DMA (ASYNCcnt),
//    double-buffered so DMA for tile i+1 overlaps WMMAs on tile i.
//  * V is repacked to [B*H, DV, S] so flash-attention PV fragments are
//    contiguous too (the repack costs ~3us of HBM time).
// ---------------------------------------------------------------------------

#define S_LEN   2048
#define BATCH   2
#define NHEADS  16
#define DMODEL  2048
#define KVRANK  512
#define DNOPE   128
#define DROPE   64
#define DV      128
// derived
#define ROWS    (BATCH * S_LEN)            // 4096
#define QCOLS   (NHEADS * (DNOPE + DROPE)) // 3072
#define KVACOLS (KVRANK + DROPE)           // 576
#define KVCOLS  (NHEADS * (DNOPE + DV))    // 4096
#define OCOLS   (NHEADS * DV)              // 2048

typedef __attribute__((ext_vector_type(16))) _Float16 v16h;
typedef __attribute__((ext_vector_type(8)))  float    v8f;

__device__ __forceinline__ v8f zero8() {
  v8f z = {0.f, 0.f, 0.f, 0.f, 0.f, 0.f, 0.f, 0.f};
  return z;
}

__device__ __forceinline__ v8f wmma_f16(v16h a, v16h b, v8f c) {
  // 8 args: (neg_a, A, neg_b, B, c_mod, C, reuse_a, reuse_b)
  return __builtin_amdgcn_wmma_f32_16x16x32_f16(false, a, false, b,
                                                (short)0, c, false, false);
}

// --- gfx1250 async direct-to-LDS copy (ASYNCcnt path) -----------------------
// LDS byte offset == low 32 bits of the generic shared pointer (aperture rule).
__device__ __forceinline__ void async_g2l_b128(_Float16* lds,
                                               const _Float16* gptr) {
  const unsigned int laddr = (unsigned int)(unsigned long long)lds;
  const unsigned long long ga = (unsigned long long)gptr;
  asm volatile("global_load_async_to_lds_b128 %0, %1, off"
               :
               : "v"(laddr), "v"(ga)
               : "memory");
}

__device__ __forceinline__ void wait_async0() {
#if defined(__has_builtin)
#if __has_builtin(__builtin_amdgcn_s_wait_asynccnt)
  __builtin_amdgcn_s_wait_asynccnt(0);
#else
  asm volatile("s_wait_asynccnt 0" ::: "memory");
#endif
#else
  asm volatile("s_wait_asynccnt 0" ::: "memory");
#endif
}

// --- fragment loaders per ISA 7.12.2 layouts (wave32) -----------------------

// A: 16x32 (MxK) f16. lanes 0-15 -> M=lane; lanes 16-31 -> M=lane-16, K+=8.
// VGPR j<4: K=2j,2j+1 ; VGPR j>=4: K=2j+8,2j+9.
// Per lane this is two contiguous 8-half runs -> 2 x b128.
__device__ __forceinline__ v16h load_a_frag(const _Float16* p, int lda) {
  const int lane = threadIdx.x & 31;
  const int m    = lane & 15;
  const int koff = (lane >> 4) << 3;     // 0 or 8
  v16h a;
#pragma unroll
  for (int j = 0; j < 8; ++j) {
    const int kb = ((j < 4) ? (2 * j) : (2 * j + 8)) + koff;
    a[2 * j]     = p[(size_t)m * lda + kb];
    a[2 * j + 1] = p[(size_t)m * lda + kb + 1];
  }
  return a;
}

// B: 32x16 (KxN) fragment from N-major storage (stride ldn between N-rows):
// lanes 0-15: N=lane, K=0..7,16..23 ; lanes 16-31: N=lane-16, K=8..15,24..31.
// Per lane: two contiguous 8-half runs -> 2 x b128.
__device__ __forceinline__ v16h load_b_frag_nk(const _Float16* p, int ldn) {
  const int lane = threadIdx.x & 31;
  const int n    = lane & 15;
  const int kh   = (lane >> 4) << 3;
  v16h b;
#pragma unroll
  for (int i = 0; i < 16; ++i) b[i] = p[(size_t)n * ldn + kh + (i & 7) + ((i >> 3) << 4)];
  return b;
}

// --- fp32 -> f16 downconvert (row-major, for the activation x) --------------

__global__ void f32_to_f16_kernel(const float* __restrict__ src,
                                  _Float16* __restrict__ dst, size_t n) {
  size_t i = (size_t)blockIdx.x * blockDim.x + threadIdx.x;
  if (i < n) dst[i] = (_Float16)src[i];
}

// --- fp32 [K,N] -> f16 [N,K] convert+transpose (weights) --------------------
// 32x32 LDS tiles, block (32,8): both global read and write fully coalesced.

__global__ void cvt_transpose_kernel(const float* __restrict__ src,
                                     _Float16* __restrict__ dst,
                                     int K, int N) {
  __shared__ _Float16 t[32][33];
  const int k0 = blockIdx.y * 32, n0 = blockIdx.x * 32;
#pragma unroll
  for (int i = 0; i < 4; ++i) {
    const int kk = threadIdx.y + i * 8;
    t[kk][threadIdx.x] =
        (_Float16)src[(size_t)(k0 + kk) * N + n0 + threadIdx.x];
  }
  __syncthreads();
#pragma unroll
  for (int i = 0; i < 4; ++i) {
    const int nn = threadIdx.y + i * 8;
    dst[(size_t)(n0 + nn) * K + k0 + threadIdx.x] = t[threadIdx.x][nn];
  }
}

// --- V repack: kv_raw [ROWS, 4096] (per head [k_nope|v]) -> vT [B*H, DV, S] -

__global__ void transpose_v_kernel(const _Float16* __restrict__ kv,
                                   _Float16* __restrict__ vT) {
  __shared__ _Float16 t[32][33];
  const int bh = blockIdx.z;
  const int b  = bh >> 4, h = bh & 15;
  const int s0 = blockIdx.x * 32, d0 = blockIdx.y * 32;
  const size_t src0 =
      ((size_t)b * S_LEN + s0) * KVCOLS + h * (DNOPE + DV) + DNOPE + d0;
#pragma unroll
  for (int i = 0; i < 4; ++i) {
    const int ss = threadIdx.y + i * 8;
    t[ss][threadIdx.x] = kv[src0 + (size_t)ss * KVCOLS + threadIdx.x];
  }
  __syncthreads();
  const size_t dst0 = ((size_t)bh * DV + d0) * S_LEN + s0;
#pragma unroll
  for (int i = 0; i < 4; ++i) {
    const int dd = threadIdx.y + i * 8;
    vT[dst0 + (size_t)dd * S_LEN + threadIdx.x] = t[threadIdx.x][dd];
  }
}

// --- WMMA GEMM: C[M,N] = A[M,K] x BT[N,K]^T, f16 in, f32 acc ----------------
// BM=128, BN=64, BK=32. 128 threads = 4 waves; wave w owns rows w*32..w*32+31
// and all 64 columns -> 2x4 fragments, 8 WMMAs per k-step per wave.
// A tile (128x32, K-major) and B tile (64x32, N-major) staged via async
// direct-to-LDS DMA, double buffered: DMA for tile i+1 overlaps tile i WMMAs.

template <typename OutT>
__global__ __launch_bounds__(128) void gemm_f16_wmma(
    const _Float16* __restrict__ A, const _Float16* __restrict__ BT,
    OutT* __restrict__ C, int K, int lda, int ldb, int ldc) {
  __shared__ _Float16 As[2][128 * 32];   // 16 KB
  __shared__ _Float16 Bs[2][64 * 32];    // 8 KB, N-major (ld = 32)
  const int tid  = threadIdx.x;
  const int wave = tid >> 5;
  const int lane = tid & 31;
  const size_t bm = (size_t)blockIdx.y * 128;
  const size_t bn = (size_t)blockIdx.x * 64;
  const int wm = wave * 32;

  v8f acc[2][4];
#pragma unroll
  for (int i = 0; i < 2; ++i)
#pragma unroll
    for (int j = 0; j < 4; ++j) acc[i][j] = zero8();

  // staging assignment (per thread):
  //   A tile 128x32: thread t copies row t        (32 halves = 4 x b128)
  //   B tile  64x32: thread t copies n-row t>>1,
  //                  k-cols (t&1)*16..            (16 halves = 2 x b128)
  const int brow = tid >> 1, bcol = (tid & 1) * 16;

  auto stage = [&](int buf, int k0) {
    const _Float16* ga = A + (bm + tid) * (size_t)lda + k0;
    _Float16* la = As[buf] + tid * 32;
    async_g2l_b128(la + 0,  ga + 0);
    async_g2l_b128(la + 8,  ga + 8);
    async_g2l_b128(la + 16, ga + 16);
    async_g2l_b128(la + 24, ga + 24);
    const _Float16* gb = BT + (bn + brow) * (size_t)ldb + k0 + bcol;
    _Float16* lb = Bs[buf] + brow * 32 + bcol;
    async_g2l_b128(lb + 0, gb + 0);
    async_g2l_b128(lb + 8, gb + 8);
  };

  stage(0, 0);
  for (int k0 = 0; k0 < K; k0 += 32) {
    const int buf = (k0 >> 5) & 1;
    wait_async0();          // own DMA done
    __syncthreads();        // everyone's DMA done; prior tile reads retired
    if (k0 + 32 < K) stage(buf ^ 1, k0 + 32);

    v16h a0 = load_a_frag(As[buf] + (wm + 0) * 32, 32);
    v16h a1 = load_a_frag(As[buf] + (wm + 16) * 32, 32);
    v16h b0 = load_b_frag_nk(Bs[buf] + 0 * 32, 32);
    v16h b1 = load_b_frag_nk(Bs[buf] + 16 * 32, 32);
    v16h b2 = load_b_frag_nk(Bs[buf] + 32 * 32, 32);
    v16h b3 = load_b_frag_nk(Bs[buf] + 48 * 32, 32);
    acc[0][0] = wmma_f16(a0, b0, acc[0][0]);
    acc[0][1] = wmma_f16(a0, b1, acc[0][1]);
    acc[0][2] = wmma_f16(a0, b2, acc[0][2]);
    acc[0][3] = wmma_f16(a0, b3, acc[0][3]);
    acc[1][0] = wmma_f16(a1, b0, acc[1][0]);
    acc[1][1] = wmma_f16(a1, b1, acc[1][1]);
    acc[1][2] = wmma_f16(a1, b2, acc[1][2]);
    acc[1][3] = wmma_f16(a1, b3, acc[1][3]);
  }

  // C fragment: VGPR r -> row (mo + r), lane half selects rows 0-7 / 8-15.
  const int n  = lane & 15;
  const int mo = (lane >> 4) * 8;
#pragma unroll
  for (int i = 0; i < 2; ++i) {
#pragma unroll
    for (int r = 0; r < 8; ++r) {
      const size_t row = (bm + wm + i * 16 + mo + r) * (size_t)ldc;
#pragma unroll
      for (int j = 0; j < 4; ++j)
        C[row + bn + j * 16 + n] = (OutT)acc[i][j][r];
    }
  }
}

// --- RoPE (in place, f32 math on f16 data) ----------------------------------

__global__ void rope_q_kernel(_Float16* __restrict__ q) {
  const int idx = blockIdx.x * blockDim.x + threadIdx.x;  // (row, h, j)
  if (idx >= ROWS * NHEADS * 32) return;
  const int j   = idx & 31;
  const int h   = (idx >> 5) & 15;
  const int row = idx >> 9;
  const int s   = row & (S_LEN - 1);
  _Float16* p = q + (size_t)row * QCOLS + h * (DNOPE + DROPE) + DNOPE;
  const float f = (float)s * __powf(10000.0f, -(float)j * (1.0f / 32.0f));
  float sn, cs;
  __sincosf(f, &sn, &cs);
  const float x1 = (float)p[j], x2 = (float)p[j + 32];
  p[j]      = (_Float16)(x1 * cs - x2 * sn);
  p[j + 32] = (_Float16)(x2 * cs + x1 * sn);
}

__global__ void rope_k_kernel(_Float16* __restrict__ kva) {
  const int idx = blockIdx.x * blockDim.x + threadIdx.x;  // (row, j)
  if (idx >= ROWS * 32) return;
  const int j   = idx & 31;
  const int row = idx >> 5;
  const int s   = row & (S_LEN - 1);
  _Float16* p = kva + (size_t)row * KVACOLS + KVRANK;
  const float f = (float)s * __powf(10000.0f, -(float)j * (1.0f / 32.0f));
  float sn, cs;
  __sincosf(f, &sn, &cs);
  const float x1 = (float)p[j], x2 = (float)p[j + 32];
  p[j]      = (_Float16)(x1 * cs - x2 * sn);
  p[j + 32] = (_Float16)(x2 * cs + x1 * sn);
}

// --- Flash attention --------------------------------------------------------
// grid = (S/64, B*H), block = 128 (4 waves). Wave w owns 16 query rows.
// Q frags (d=192 -> 6 A-frags) live in registers for the whole key loop.
// Per 32-key chunk: 12 K frags (contiguous b128 loads) -> 12 score WMMAs;
// 8 V frags prefetched from the transposed vT buffer (contiguous b128 loads)
// before the softmax so VALU work hides the latency; then 8 PV WMMAs.

__global__ __launch_bounds__(128) void mla_attention_kernel(
    const _Float16* __restrict__ q,    // [ROWS, 3072] (rope applied)
    const _Float16* __restrict__ kv,   // [ROWS, 4096] = per head [k_nope|v]
    const _Float16* __restrict__ kr,   // [ROWS, 576], cols 512.. = roped k_rope
    const _Float16* __restrict__ vT,   // [B*H, DV, S]
    _Float16* __restrict__ o) {        // [ROWS, 2048]
  __shared__ _Float16 pst[4][16 * 32];
  const int wave = threadIdx.x >> 5;
  const int lane = threadIdx.x & 31;
  const int b = blockIdx.y >> 4;
  const int h = blockIdx.y & 15;
  const int q0 = blockIdx.x * 64 + wave * 16;
  const size_t row0 = (size_t)b * S_LEN;
  _Float16* pl = pst[wave];

  v16h qa[6];
  {
    const _Float16* qp = q + (row0 + q0) * (size_t)QCOLS + h * (DNOPE + DROPE);
#pragma unroll
    for (int c = 0; c < 6; ++c) qa[c] = load_a_frag(qp + c * 32, QCOLS);
  }

  v8f oacc[8];
#pragma unroll
  for (int j = 0; j < 8; ++j) oacc[j] = zero8();
  float rmax[8], rsum[8];
#pragma unroll
  for (int r = 0; r < 8; ++r) { rmax[r] = -3.0e38f; rsum[r] = 0.f; }

  const float scale = 0.07216878364870323f;  // 1/sqrt(192)
  const int n  = lane & 15;
  const int mo = (lane >> 4) * 8;
  const int kend = q0 + 16;                  // causal bound (exclusive)
  const _Float16* vbase = vT + (size_t)blockIdx.y * DV * S_LEN;

  for (int kb = 0; kb < kend; kb += 32) {
    const _Float16* kbase = kv + (row0 + kb) * (size_t)KVCOLS + h * (DNOPE + DV);
    const _Float16* rbase = kr + (row0 + kb) * (size_t)KVACOLS + KVRANK;

    // prefetch next chunk's K rows into GL2 while this chunk is processed
    if (kb + 32 < kend) {
      __builtin_prefetch(kbase + (size_t)32 * KVCOLS, 0, 1);
      __builtin_prefetch(rbase + (size_t)32 * KVACOLS, 0, 1);
    }

    // all 12 K fragments first -> single load burst, then 12 dense WMMAs
    v16h kf[12];
#pragma unroll
    for (int c = 0; c < 4; ++c) {
      kf[2 * c]     = load_b_frag_nk(kbase + c * 32, KVCOLS);
      kf[2 * c + 1] = load_b_frag_nk(kbase + (size_t)16 * KVCOLS + c * 32, KVCOLS);
    }
#pragma unroll
    for (int c = 0; c < 2; ++c) {
      kf[8 + 2 * c]     = load_b_frag_nk(rbase + c * 32, KVACOLS);
      kf[8 + 2 * c + 1] = load_b_frag_nk(rbase + (size_t)16 * KVACOLS + c * 32, KVACOLS);
    }

    v8f s0 = zero8(), s1 = zero8();
#pragma unroll
    for (int c = 0; c < 4; ++c) {
      s0 = wmma_f16(qa[c], kf[2 * c], s0);
      s1 = wmma_f16(qa[c], kf[2 * c + 1], s1);
    }
#pragma unroll
    for (int c = 0; c < 2; ++c) {
      s0 = wmma_f16(qa[4 + c], kf[8 + 2 * c], s0);
      s1 = wmma_f16(qa[4 + c], kf[8 + 2 * c + 1], s1);
    }

    // prefetch V fragments (contiguous in key from vT) into registers;
    // softmax VALU work hides the latency
    v16h vf[8];
#pragma unroll
    for (int j = 0; j < 8; ++j)
      vf[j] = load_b_frag_nk(vbase + (size_t)(j * 16) * S_LEN + kb, S_LEN);

    // scale + causal mask (select, not branch: EXEC stays full for WMMA)
    float sv0[8], sv1[8];
#pragma unroll
    for (int r = 0; r < 8; ++r) {
      const int qrow = q0 + mo + r;
      sv0[r] = (kb + n      <= qrow) ? s0[r] * scale : -3.0e38f;
      sv1[r] = (kb + 16 + n <= qrow) ? s1[r] * scale : -3.0e38f;
    }

    // online softmax; rows live in 16-lane groups -> shfl_xor width-16 tree
#pragma unroll
    for (int r = 0; r < 8; ++r) {
      float m = fmaxf(sv0[r], sv1[r]);
#pragma unroll
      for (int off = 1; off < 16; off <<= 1) m = fmaxf(m, __shfl_xor(m, off, 32));
      const float nm   = fmaxf(rmax[r], m);
      const float corr = __expf(rmax[r] - nm);
      const float p0 = __expf(sv0[r] - nm);
      const float p1 = __expf(sv1[r] - nm);
      float rs = p0 + p1;
#pragma unroll
      for (int off = 1; off < 16; off <<= 1) rs += __shfl_xor(rs, off, 32);
      rsum[r] = rsum[r] * corr + rs;
      rmax[r] = nm;
#pragma unroll
      for (int j = 0; j < 8; ++j) oacc[j][r] *= corr;
      // stage P (C-fragment layout) into LDS so it can be re-read as A-frag
      pl[(mo + r) * 32 + n]      = (_Float16)p0;
      pl[(mo + r) * 32 + 16 + n] = (_Float16)p1;
    }

    // P(16x32) x V(32x16) for each of the 8 v-dim chunks
    v16h pa = load_a_frag(pl, 32);
#pragma unroll
    for (int j = 0; j < 8; ++j) oacc[j] = wmma_f16(pa, vf[j], oacc[j]);
  }

  // normalize + store to [ROWS, NHEADS*DV]
#pragma unroll
  for (int r = 0; r < 8; ++r) {
    const float inv = 1.0f / rsum[r];
    const size_t orow = (row0 + q0 + mo + r) * (size_t)OCOLS + h * DV;
#pragma unroll
    for (int j = 0; j < 8; ++j)
      o[orow + j * 16 + n] = (_Float16)(oacc[j][r] * inv);
  }
}

// ---------------------------------------------------------------------------

extern "C" void kernel_launch(void* const* d_in, const int* in_sizes, int n_in,
                              void* d_out, int out_size, void* d_ws,
                              size_t ws_size, hipStream_t stream) {
  (void)in_sizes; (void)n_in; (void)out_size; (void)ws_size;
  const float* x    = (const float*)d_in[0];
  const float* Wq   = (const float*)d_in[1];
  const float* Wkva = (const float*)d_in[2];
  const float* Wkvb = (const float*)d_in[3];
  const float* Wo   = (const float*)d_in[4];
  float* out = (float*)d_out;

  char* ws = (char*)d_ws;
  size_t off = 0;
  auto halloc = [&](size_t elems) -> _Float16* {
    _Float16* p = (_Float16*)(ws + off);
    off += ((elems * sizeof(_Float16)) + 255) & ~(size_t)255;
    return p;
  };
  _Float16* xh     = halloc((size_t)ROWS * DMODEL);     // 16 MB
  _Float16* WqT    = halloc((size_t)DMODEL * QCOLS);    // 12.6 MB  [N,K]
  _Float16* WkvaT  = halloc((size_t)DMODEL * KVACOLS);  // 2.4 MB   [N,K]
  _Float16* WkvbT  = halloc((size_t)KVRANK * KVCOLS);   // 4.2 MB   [N,K]
  _Float16* WoT    = halloc((size_t)OCOLS * DMODEL);    // 8.4 MB   [N,K]
  _Float16* q_raw  = halloc((size_t)ROWS * QCOLS);      // 25.2 MB
  _Float16* kva    = halloc((size_t)ROWS * KVACOLS);    // 4.7 MB
  _Float16* kv_raw = halloc((size_t)ROWS * KVCOLS);     // 33.6 MB
  _Float16* vTb    = halloc((size_t)BATCH * NHEADS * DV * S_LEN); // 16.8 MB
  _Float16* oh     = halloc((size_t)ROWS * OCOLS);      // 16.8 MB

  // activation: row-major convert
  {
    const size_t n = (size_t)ROWS * DMODEL;
    f32_to_f16_kernel<<<dim3((unsigned)((n + 255) / 256)), 256, 0, stream>>>(
        x, xh, n);
  }
  // weights: convert + transpose to N-major
  auto cvtT = [&](const float* s, _Float16* d, int K, int N) {
    cvt_transpose_kernel<<<dim3(N / 32, K / 32), dim3(32, 8), 0, stream>>>(
        s, d, K, N);
  };
  cvtT(Wq,   WqT,   DMODEL, QCOLS);
  cvtT(Wkva, WkvaT, DMODEL, KVACOLS);
  cvtT(Wkvb, WkvbT, KVRANK, KVCOLS);
  cvtT(Wo,   WoT,   OCOLS,  DMODEL);

  // q = x @ Wq                [4096,2048] x [2048,3072]
  gemm_f16_wmma<_Float16><<<dim3(QCOLS / 64, ROWS / 128), 128, 0, stream>>>(
      xh, WqT, q_raw, DMODEL, DMODEL, DMODEL, QCOLS);
  // kv_a = x @ Wkv_a          [4096,2048] x [2048,576]
  gemm_f16_wmma<_Float16><<<dim3(KVACOLS / 64, ROWS / 128), 128, 0, stream>>>(
      xh, WkvaT, kva, DMODEL, DMODEL, DMODEL, KVACOLS);
  // RoPE on q_rope (per head) and k_rope (shared)
  rope_q_kernel<<<dim3((ROWS * NHEADS * 32) / 256), 256, 0, stream>>>(q_raw);
  rope_k_kernel<<<dim3((ROWS * 32) / 256), 256, 0, stream>>>(kva);
  // kv = c_kv @ Wkv_b         [4096,512] (lda=576) x [512,4096]
  gemm_f16_wmma<_Float16><<<dim3(KVCOLS / 64, ROWS / 128), 128, 0, stream>>>(
      kva, WkvbT, kv_raw, KVRANK, KVACOLS, KVRANK, KVCOLS);
  // repack V to [B*H, DV, S] for contiguous PV fragments
  transpose_v_kernel<<<dim3(S_LEN / 32, DV / 32, BATCH * NHEADS), dim3(32, 8),
                       0, stream>>>(kv_raw, vTb);
  // flash attention
  mla_attention_kernel<<<dim3(S_LEN / 64, BATCH * NHEADS), 128, 0, stream>>>(
      q_raw, kv_raw, kva, vTb, oh);
  // out = attn_out @ Wo       [4096,2048] x [2048,2048] -> fp32 d_out
  gemm_f16_wmma<float><<<dim3(DMODEL / 64, ROWS / 128), 128, 0, stream>>>(
      oh, WoT, out, OCOLS, OCOLS, OCOLS, DMODEL);
}
